// MultiHeadAttention_20976620273714
// MI455X (gfx1250) — compile-verified, matrix operand order checked by reading, not run on hardware
//
#include <hip/hip_runtime.h>
#include <cstdint>

typedef __bf16 bf16;
typedef __attribute__((ext_vector_type(16))) __bf16 v16bf;
typedef __attribute__((ext_vector_type(8)))  __bf16 bf16x8;
typedef __attribute__((ext_vector_type(8)))  float  v8f;
typedef __attribute__((ext_vector_type(4)))  unsigned int u32x4;
typedef __attribute__((ext_vector_type(8)))  int i32x8;
typedef __attribute__((ext_vector_type(4)))  int i32x4;

#define LOG2E 1.4426950408889634f

// ---------- helpers ----------

__device__ __forceinline__ bf16 f2bf(float f) {
  unsigned u = __builtin_bit_cast(unsigned, f);
  unsigned r = (u + 0x7FFFu + ((u >> 16) & 1u)) >> 16;   // round-to-nearest-even
  unsigned short s = (unsigned short)r;
  return __builtin_bit_cast(bf16, s);
}

__device__ __forceinline__ v8f zero_v8f() {
  v8f z;
#pragma unroll
  for (int i = 0; i < 8; ++i) z[i] = 0.0f;
  return z;
}

// ds_swizzle xor-shuffle within 16-lane groups (group-of-32 mode:
// offset = xor_mask<<10 | or_mask<<5 | and_mask; and_mask=0x1f).
#define SWZ_XOR(v, m) \
  __builtin_bit_cast(float, __builtin_amdgcn_ds_swizzle( \
      __builtin_bit_cast(int, (v)), ((m) << 10) | 0x1F))

__device__ __forceinline__ float rowmax16(float v) {
  v = fmaxf(v, SWZ_XOR(v, 8));
  v = fmaxf(v, SWZ_XOR(v, 4));
  v = fmaxf(v, SWZ_XOR(v, 2));
  v = fmaxf(v, SWZ_XOR(v, 1));
  return v;
}

__device__ __forceinline__ float rowsum16(float v) {
  v += SWZ_XOR(v, 8);
  v += SWZ_XOR(v, 4);
  v += SWZ_XOR(v, 2);
  v += SWZ_XOR(v, 1);
  return v;
}

// Async copy 16 bytes global -> LDS (ASYNCcnt path). vdst VGPR holds the
// per-lane LDS byte offset (= low 32 bits of the generic pointer).
__device__ __forceinline__ void async_ld_b128(const bf16* lds_ptr, const bf16* gptr) {
  unsigned loff = (unsigned)(uintptr_t)lds_ptr;
  unsigned long long ga = (unsigned long long)(uintptr_t)gptr;
  asm volatile("global_load_async_to_lds_b128 %0, %1, off"
               :: "v"(loff), "v"(ga) : "memory");
}

__device__ __forceinline__ void wait_async0() {
  asm volatile("s_wait_asynccnt 0x0" ::: "memory");
}

// Tensor Data Mover: DMA a 2D tile (tile_d0 contiguous elems x tile_d1 rows,
// 2-byte elements, row stride stride0) from global into LDS at lds_off.
// D# layout per CDNA5 ISA ch.8: group0 {count, lds_addr, global_addr, type=2},
// group1 {data_size=1, tensor dims, tile dims, dim0 stride}, groups 2/3 zero.
__device__ __forceinline__ void tdm_load_2d(unsigned lds_off, const bf16* gptr,
                                            unsigned tensor_d0, unsigned tensor_d1,
                                            unsigned tile_d0, unsigned tile_d1,
                                            unsigned stride0) {
  unsigned long long ga = (unsigned long long)(uintptr_t)gptr;
  u32x4 g0;
  g0[0] = 1u;                                             // count=1 (user D#)
  g0[1] = lds_off;                                        // lds_addr (bytes)
  g0[2] = (unsigned)(ga & 0xFFFFFFFFu);                   // global_addr[31:0]
  g0[3] = (unsigned)((ga >> 32) & 0x1FFFFFFu) | (2u << 30); // [56:32] | type=2
  i32x8 g1;
  g1[0] = (int)(1u << 16);                                // data_size=1 (2B)
  g1[1] = (int)((tensor_d0 & 0xFFFFu) << 16);             // tensor_dim0[15:0]
  g1[2] = (int)(((tensor_d0 >> 16) & 0xFFFFu) | ((tensor_d1 & 0xFFFFu) << 16));
  g1[3] = (int)(((tensor_d1 >> 16) & 0xFFFFu) | ((tile_d0 & 0xFFFFu) << 16));
  g1[4] = (int)(tile_d1 & 0xFFFFu);                       // tile_dim1, tile_dim2=0
  g1[5] = (int)stride0;                                   // dim0_stride[31:0]
  g1[6] = 0;
  g1[7] = 0;
  i32x4 z4 = {0, 0, 0, 0};
  i32x8 z8 = {0, 0, 0, 0, 0, 0, 0, 0};
  __builtin_amdgcn_tensor_load_to_lds(g0, g1, z4, z4, z8, 0);
}

__device__ __forceinline__ void wait_tensor0() {
  __builtin_amdgcn_s_wait_tensorcnt((short)0);
}

// Load a 16x32 (rows x K) bf16 fragment where K is the contiguous dimension.
// ISA 16-bit A/B layout: lane<16 holds K {kb..kb+7, 16+kb..16+kb+7}, kb = 8*(lane>=16).
__device__ __forceinline__ v16bf load_frag(const bf16* base, int ld) {
  const int lane = threadIdx.x & 31;
  const bf16* p = base + (long)(lane & 15) * ld + ((lane >> 4) << 3);
  bf16x8 lo = *reinterpret_cast<const bf16x8*>(p);
  bf16x8 hi = *reinterpret_cast<const bf16x8*>(p + 16);
  v16bf f;
#pragma unroll
  for (int i = 0; i < 8; ++i) { f[i] = lo[i]; f[i + 8] = hi[i]; }
  return f;
}

__device__ __forceinline__ v8f wmma_bf16(v16bf a, v16bf b, v8f c) {
  return __builtin_amdgcn_wmma_f32_16x16x32_bf16(
      /*neg_a=*/false, a, /*neg_b=*/false, b,
      /*c_mod=*/(short)0, c, /*reuse_a=*/false, /*reuse_b=*/false);
}

// ---------- kernel 1: fp32 -> bf16 convert ----------

__global__ void f32_to_bf16_k(const float* __restrict__ in, bf16* __restrict__ out, int n) {
  int i = blockIdx.x * blockDim.x + threadIdx.x;
  if (i < n) out[i] = f2bf(in[i]);
}

// ---------- kernel 2: WMMA GEMM  C = (A @ W^T + bias) * scale ----------
// A: M x Kd bf16 row-major.  W: N x Kd bf16 row-major (so B[k][n] = W[n][k]).
// Tile staging via Tensor Data Mover (wave 0 programs two D# per k-step),
// double buffered; compute fragments come from LDS.
// MODE 0: bf16 row-major out (Q,K); MODE 1: Vt[b][h][d][q]; MODE 2: f32 out.

template <int MODE>
__global__ __launch_bounds__(256, 1)
void gemm_wmma(const bf16* __restrict__ A, const bf16* __restrict__ W,
               const float* __restrict__ bias, void* __restrict__ Cout,
               int M, int N, int Kd, float scale) {
  __shared__ __align__(16) bf16 sA[2][128 * 32];   // 2 x 8 KB
  __shared__ __align__(16) bf16 sW[2][64 * 32];    // 2 x 4 KB

  const int tid  = threadIdx.x;
  const int lane = tid & 31;
  const int wv   = tid >> 5;              // 8 waves
  const int wr   = wv & 3;                // 4 waves along M
  const int wc   = wv >> 2;               // 2 waves along N
  const int m_blk = blockIdx.x * 128;
  const int n_blk = blockIdx.y * 64;

  auto issue = [&](int buf, int k0) {
    // one TDM descriptor per tile; issued by a single wave (EXEC ignored)
    tdm_load_2d((unsigned)(uintptr_t)&sA[buf][0],
                A + (long)m_blk * Kd + k0, Kd, M, 32, 128, Kd);
    tdm_load_2d((unsigned)(uintptr_t)&sW[buf][0],
                W + (long)n_blk * Kd + k0, Kd, N, 32, 64, Kd);
  };

  v8f acc[2][2];
#pragma unroll
  for (int i = 0; i < 2; ++i)
#pragma unroll
    for (int j = 0; j < 2; ++j) acc[i][j] = zero_v8f();

  const int KT = Kd / 32;
  if (wv == 0) issue(0, 0);
  for (int kt = 0; kt < KT; ++kt) {
    const int cur = kt & 1;
    if (wv == 0) wait_tensor0();   // issuing wave's DMAs complete
    __syncthreads();               // publish tile to all waves
    if (kt + 1 < KT && wv == 0) issue(cur ^ 1, (kt + 1) * 32);

    v16bf a0 = load_frag(&sA[cur][(wr * 32) * 32], 32);
    v16bf a1 = load_frag(&sA[cur][(wr * 32 + 16) * 32], 32);
    v16bf b0 = load_frag(&sW[cur][(wc * 32) * 32], 32);
    v16bf b1 = load_frag(&sW[cur][(wc * 32 + 16) * 32], 32);
    acc[0][0] = wmma_bf16(a0, b0, acc[0][0]);
    acc[0][1] = wmma_bf16(a0, b1, acc[0][1]);
    acc[1][0] = wmma_bf16(a1, b0, acc[1][0]);
    acc[1][1] = wmma_bf16(a1, b1, acc[1][1]);
  }

  const int hi  = lane >> 4;
  const int col = lane & 15;
  const int m0  = m_blk + wr * 32;
  const int n0  = n_blk + wc * 32;
#pragma unroll
  for (int i = 0; i < 2; ++i) {
#pragma unroll
    for (int j = 0; j < 2; ++j) {
      const int mt = m0 + 16 * i;
      const int nt = n0 + 16 * j + col;
      const float bval = bias[nt];
      if (MODE == 0) {
        bf16* C = (bf16*)Cout;
#pragma unroll
        for (int r = 0; r < 8; ++r) {
          int row = mt + r + 8 * hi;
          C[(long)row * N + nt] = f2bf((acc[i][j][r] + bval) * scale);
        }
      } else if (MODE == 2) {
        float* C = (float*)Cout;
#pragma unroll
        for (int r = 0; r < 8; ++r) {
          int row = mt + r + 8 * hi;
          C[(long)row * N + nt] = acc[i][j][r] * scale + bval;
        }
      } else {
        // Vt[b][h][d][q]; 8 consecutive tokens per lane -> one 16B store
        bf16* C = (bf16*)Cout;
        const int tok = mt + 8 * hi;
        const int b = tok >> 12, q = tok & 4095;
        const int h = nt >> 6,  d = nt & 63;
        bf16x8 v;
#pragma unroll
        for (int r = 0; r < 8; ++r) v[r] = f2bf(acc[i][j][r] + bval);
        *reinterpret_cast<bf16x8*>(C + (((long)(b * 8 + h) * 64 + d) * 4096 + q)) = v;
      }
    }
  }
}

// ---------- kernel 3: flash attention ----------
// grid: (L/128, B*H), block 256 (8 waves). Each wave: 16 query rows.
// All 8 waves share the same (b,h) key block -> stage K/V once per block in
// LDS via per-lane async loads (double buffered): 8x global-traffic reuse.
// Softmax runs in the exp2 domain: Q is pre-scaled by log2(e)/sqrt(dk), so
// every exponent is a bare v_exp_f32.
// Q,K: bf16 [tok][512].  Vt: bf16 [b][h][d][key].

#define MASK_NEG (-10000.0f * LOG2E)

__global__ __launch_bounds__(256, 1)
void attn_flash(const bf16* __restrict__ Q, const bf16* __restrict__ Kb,
                const bf16* __restrict__ Vt, const int* __restrict__ mask,
                bf16* __restrict__ Obf) {
  __shared__ __align__(16) bf16 sK[2][32 * 64];     // 2 x 4 KB: [key][dk]
  __shared__ __align__(16) bf16 sV[2][64 * 32];     // 2 x 4 KB: [dk][key]
  __shared__ __align__(16) bf16 plds[8][16 * 32];   // per-wave P staging

  const int tid  = threadIdx.x;
  const int lane = tid & 31;
  const int wv   = tid >> 5;
  const int bh   = blockIdx.y;
  const int b    = bh >> 3;
  const int h    = bh & 7;
  const int q0   = blockIdx.x * 128 + wv * 16;
  const int col  = lane & 15;
  const int hi   = lane >> 4;

  const long tokq  = (long)b * 4096 + q0;
  const long vbase = (long)(b * 8 + h) * 64 * 4096;
  const long kbase = (long)b * 4096;

  auto issue_kv = [&](int buf, int kb0) {
    {   // K block: 32 keys x 64 dk = 256 x 16B chunks, 1 per thread
      const int row = tid >> 3, seg = tid & 7;       // row = key, 8 segs over dk
      async_ld_b128(&sK[buf][row * 64 + seg * 8],
                    Kb + (kbase + kb0 + row) * 512 + h * 64 + seg * 8);
    }
    {   // V block: 64 dk x 32 keys = 256 x 16B chunks, 1 per thread
      const int row = tid >> 2, seg = tid & 3;       // row = dk, 4 segs over keys
      async_ld_b128(&sV[buf][row * 32 + seg * 8],
                    Vt + vbase + (long)row * 4096 + kb0 + seg * 8);
    }
  };

  // Q fragments for this wave's 16 rows x dk=64 (two K=32 steps)
  v16bf qf0 = load_frag(Q + tokq * 512 + h * 64, 512);
  v16bf qf1 = load_frag(Q + tokq * 512 + h * 64 + 32, 512);

  float m_run[8], l_run[8];
  v8f o[4];
#pragma unroll
  for (int r = 0; r < 8; ++r) { m_run[r] = -3.0e38f; l_run[r] = 0.0f; }
#pragma unroll
  for (int t = 0; t < 4; ++t) o[t] = zero_v8f();

  bf16* pw = plds[wv];

  issue_kv(0, 0);
  for (int kb0 = 0, it = 0; kb0 < 4096; kb0 += 32, ++it) {
    const int cur = it & 1;
    wait_async0();
    __syncthreads();
    if (kb0 + 32 < 4096) issue_kv(cur ^ 1, kb0 + 32);

    // ---- S = Q @ K^T for two 16-key tiles (from LDS) ----
    v8f s[2];
#pragma unroll
    for (int t = 0; t < 2; ++t) {
      v16bf kf0 = load_frag(&sK[cur][(t * 16) * 64], 64);
      v16bf kf1 = load_frag(&sK[cur][(t * 16) * 64 + 32], 64);
      v8f z = zero_v8f();
      z = wmma_bf16(qf0, kf0, z);
      z = wmma_bf16(qf1, kf1, z);
      // padding mask: masked key column -> score = -10000 (log2 domain)
      const int key = kb0 + t * 16 + col;
      if (mask[kbase + key] == 0) {
#pragma unroll
        for (int r = 0; r < 8; ++r) z[r] = MASK_NEG;
      }
      s[t] = z;
    }

    // ---- online softmax (exp2 domain) over these 32 keys ----
    float alpha[8];
#pragma unroll
    for (int r = 0; r < 8; ++r) {
      const float mx   = rowmax16(fmaxf(s[0][r], s[1][r]));
      const float mnew = fmaxf(m_run[r], mx);
      const float a    = exp2f(m_run[r] - mnew);
      const float p0   = exp2f(s[0][r] - mnew);
      const float p1   = exp2f(s[1][r] - mnew);
      s[0][r] = p0; s[1][r] = p1;
      const float sum  = rowsum16(p0 + p1);
      l_run[r] = l_run[r] * a + sum;
      m_run[r] = mnew;
      alpha[r] = a;
    }

    // ---- relayout P: C-fragment -> row-major LDS tile -> A-fragment ----
#pragma unroll
    for (int t = 0; t < 2; ++t)
#pragma unroll
      for (int r = 0; r < 8; ++r)
        pw[(r + 8 * hi) * 32 + t * 16 + col] = f2bf(s[t][r]);
    // same-wave DS ops are in order; compiler inserts the dscnt wait
    v16bf pf = load_frag(pw, 32);

    // ---- O = O*alpha + P @ V (V from LDS) ----
#pragma unroll
    for (int t4 = 0; t4 < 4; ++t4) {
#pragma unroll
      for (int r = 0; r < 8; ++r) o[t4][r] *= alpha[r];
      v16bf vf = load_frag(&sV[cur][(t4 * 16) * 32], 32);
      o[t4] = wmma_bf16(pf, vf, o[t4]);
    }
  }

  // ---- epilogue: normalize (1 rcp per row) and store bf16 ----
  float linv[8];
#pragma unroll
  for (int r = 0; r < 8; ++r) linv[r] = __builtin_amdgcn_rcpf(l_run[r]);
#pragma unroll
  for (int t4 = 0; t4 < 4; ++t4)
#pragma unroll
    for (int r = 0; r < 8; ++r) {
      const int row = q0 + r + 8 * hi;
      Obf[((long)b * 4096 + row) * 512 + h * 64 + t4 * 16 + col] =
          f2bf(o[t4][r] * linv[r]);
    }
}

// ---------- launcher ----------

extern "C" void kernel_launch(void* const* d_in, const int* in_sizes, int n_in,
                              void* d_out, int out_size, void* d_ws, size_t ws_size,
                              hipStream_t stream) {
  const float* x    = (const float*)d_in[0];
  const int*   mask = (const int*)d_in[1];
  const float* Wq   = (const float*)d_in[2];
  const float* bq   = (const float*)d_in[3];
  const float* Wk   = (const float*)d_in[4];
  const float* bk   = (const float*)d_in[5];
  const float* Wv   = (const float*)d_in[6];
  const float* bv   = (const float*)d_in[7];
  const float* Wo   = (const float*)d_in[8];
  const float* bo   = (const float*)d_in[9];
  float* out = (float*)d_out;

  const long NTOK = 2L * 4096;        // B*L = 8192
  const long XE   = NTOK * 512;       // 4,194,304 elements
  const long WE   = 512L * 512;       // 262,144 elements

  bf16* xbf  = (bf16*)d_ws;
  bf16* wqbf = xbf  + XE;
  bf16* wkbf = wqbf + WE;
  bf16* wvbf = wkbf + WE;
  bf16* wobf = wvbf + WE;
  bf16* Qbf  = wobf + WE;
  bf16* Kbf  = Qbf  + XE;
  bf16* Vt   = Kbf  + XE;             // [b][h][d][key]
  bf16* Abf  = Vt   + XE;             // attention output, [tok][512]

  // 1) converts
  {
    int n = (int)XE;
    f32_to_bf16_k<<<dim3((n + 255) / 256), dim3(256), 0, stream>>>(x, xbf, n);
    int nw = (int)WE;
    dim3 gw((nw + 255) / 256);
    f32_to_bf16_k<<<gw, dim3(256), 0, stream>>>(Wq, wqbf, nw);
    f32_to_bf16_k<<<gw, dim3(256), 0, stream>>>(Wk, wkbf, nw);
    f32_to_bf16_k<<<gw, dim3(256), 0, stream>>>(Wv, wvbf, nw);
    f32_to_bf16_k<<<gw, dim3(256), 0, stream>>>(Wo, wobf, nw);
  }

  // 2) projections (Q scaled by log2(e)/sqrt(dk) for exp2-domain softmax)
  dim3 gg(8192 / 128, 512 / 64);   // (64, 8)
  gemm_wmma<0><<<gg, dim3(256), 0, stream>>>(xbf, wqbf, bq, (void*)Qbf, 8192, 512, 512,
                                             0.125f * LOG2E);
  gemm_wmma<0><<<gg, dim3(256), 0, stream>>>(xbf, wkbf, bk, (void*)Kbf, 8192, 512, 512, 1.0f);
  gemm_wmma<1><<<gg, dim3(256), 0, stream>>>(xbf, wvbf, bv, (void*)Vt,  8192, 512, 512, 1.0f);

  // 3) flash attention (8 waves / block, 128 queries / block)
  attn_flash<<<dim3(4096 / 128, 16), dim3(256), 0, stream>>>(Qbf, Kbf, Vt, mask, Abf);

  // 4) output projection (f32 out)
  gemm_wmma<2><<<gg, dim3(256), 0, stream>>>(Abf, wobf, bo, (void*)out, 8192, 512, 512, 1.0f);

  (void)in_sizes; (void)n_in; (void)out_size; (void)ws_size;
}
